// CAPGNN_11416023073170
// MI455X (gfx1250) — matrix operand order repaired and checked by reading, not compiled, not run on hardware
//
#include <hip/hip_runtime.h>
#include <hip/hip_bf16.h>
#include <math.h>

// ---------------- problem constants (match reference) ----------------
#define NNODES 100000
#define NEDGES 1600000
#define ETOT   (NEDGES + NNODES)   // edges + self loops
#define FDIM   128
#define H1DIM  128
#define DDIM   64
#define AUDIM  64
#define TITER  10
#define ALPHA  0.1f
#define BETA   0.3f

typedef __attribute__((ext_vector_type(16))) __bf16 v16bf;
typedef __attribute__((ext_vector_type(8)))  float  v8f;

__device__ __forceinline__ v8f wmma_bf16(v16bf a, v16bf b, v8f c) {
  // (neg_a, A, neg_b, B, c_mod, C, reuse_a, reuse_b)
  return __builtin_amdgcn_wmma_f32_16x16x32_bf16(false, a, false, b,
                                                 (short)0, c, false, false);
}

// ordered-uint encoding for float atomicMax
__device__ __forceinline__ unsigned enc_f32(float f) {
  unsigned u = __float_as_uint(f);
  return (u & 0x80000000u) ? ~u : (u | 0x80000000u);
}
__device__ __forceinline__ float dec_f32(unsigned u) {
  return __uint_as_float((u & 0x80000000u) ? (u & 0x7FFFFFFFu) : ~u);
}

// ---------------- weight prep: f32 [Kin][Nout] -> bf16 [Nout][Kin] ----------------
__global__ void wprep_kernel(const float* __restrict__ W, __bf16* __restrict__ Wt,
                             int Kin, int Nout) {
  int i = blockIdx.x * blockDim.x + threadIdx.x;
  if (i < Kin * Nout) {
    int k = i / Nout, n = i % Nout;
    Wt[n * Kin + k] = (__bf16)W[i];
  }
}

// ---------------- fused node MLP: Q, K, V via bf16 WMMA ----------------
// per wave: one 16-row tile. A operand lives in registers, H staged via LDS.
__global__ __launch_bounds__(256) void node_mlp_kernel(
    const float* __restrict__ x,                       // [N,128] f32
    const __bf16* __restrict__ W1t, const float* __restrict__ b1,  // [128][128]
    const __bf16* __restrict__ W2t, const float* __restrict__ b2,  // [64][128]
    const __bf16* __restrict__ Wqt, const float* __restrict__ bq,  // [64][128]
    const __bf16* __restrict__ Wkt, const float* __restrict__ bk,  // [64][128]
    float* __restrict__ Q, float* __restrict__ Kf, float* __restrict__ V,
    int numTiles)
{
  __shared__ __bf16 ldsH[8][16 * H1DIM];   // 8 waves * 4KB = 32KB

  const int wave = threadIdx.x >> 5;
  const int lane = threadIdx.x & 31;
  const int hi   = lane >> 4;     // 0: lanes 0-15, 1: lanes 16-31
  const int l16  = lane & 15;
  const int tile = blockIdx.x * 8 + wave;
  if (tile >= numTiles) return;   // wave-uniform: EXEC all-ones below
  const int rowbase = tile * 16;

  // --- A operand (x tile, 16x128) f32 -> bf16, 4 K-slabs of 32 ---
  // lane holds row (rowbase + l16); elems 0..7 = K s*32+hi*8+i, 8..15 = +16
  v16bf ax[4];
  const float* xrow = x + (size_t)(rowbase + l16) * FDIM;
#pragma unroll
  for (int s = 0; s < 4; ++s) {
    const int k0 = s * 32 + hi * 8;
#pragma unroll
    for (int i = 0; i < 8; ++i) {
      ax[s][i]     = (__bf16)xrow[k0 + i];
      ax[s][i + 8] = (__bf16)xrow[k0 + 16 + i];
    }
  }

  // --- B operand loader: Wt is [Nout][Kin] bf16; lane = column, elems 0..7 = K s*32+hi*8+i, 8..15 = +16 ---
  auto loadB = [&](const __bf16* Wt, int nt, int s) -> v16bf {
    const __bf16* p = Wt + (size_t)(nt * 16 + l16) * FDIM + s * 32 + hi * 8;
    v16bf b;
#pragma unroll
    for (int i = 0; i < 8; ++i) { b[i] = p[i]; b[i + 8] = p[16 + i]; }
    return b;
  };

  // ---------- Q = relu(x @ Wq + bq) : 4 N-tiles ----------
#pragma unroll
  for (int nt = 0; nt < 4; ++nt) {
    v8f acc = {};
#pragma unroll
    for (int s = 0; s < 4; ++s) acc = wmma_bf16(ax[s], loadB(Wqt, nt, s), acc);
    const float bias = bq[nt * 16 + l16];
#pragma unroll
    for (int v = 0; v < 8; ++v) {
      const int m = v + hi * 8;
      Q[(size_t)(rowbase + m) * AUDIM + nt * 16 + l16] = fmaxf(acc[v] + bias, 0.0f);
    }
  }

  // ---------- K = relu(x @ Wk + bk) ----------
#pragma unroll
  for (int nt = 0; nt < 4; ++nt) {
    v8f acc = {};
#pragma unroll
    for (int s = 0; s < 4; ++s) acc = wmma_bf16(ax[s], loadB(Wkt, nt, s), acc);
    const float bias = bk[nt * 16 + l16];
#pragma unroll
    for (int v = 0; v < 8; ++v) {
      const int m = v + hi * 8;
      Kf[(size_t)(rowbase + m) * AUDIM + nt * 16 + l16] = fmaxf(acc[v] + bias, 0.0f);
    }
  }

  // ---------- H = relu(x @ W1 + b1) -> LDS (bf16) : 8 N-tiles ----------
  __bf16* hl = &ldsH[wave][0];
#pragma unroll
  for (int nt = 0; nt < 8; ++nt) {
    v8f acc = {};
#pragma unroll
    for (int s = 0; s < 4; ++s) acc = wmma_bf16(ax[s], loadB(W1t, nt, s), acc);
    const float bias = b1[nt * 16 + l16];
#pragma unroll
    for (int v = 0; v < 8; ++v) {
      const int m = v + hi * 8;
      hl[m * H1DIM + nt * 16 + l16] = (__bf16)fmaxf(acc[v] + bias, 0.0f);
    }
  }

  // ---------- reload H tile as A operand from LDS ----------
  v16bf ah[4];
  const __bf16* hrow = hl + l16 * H1DIM;
#pragma unroll
  for (int s = 0; s < 4; ++s) {
    const int k0 = s * 32 + hi * 8;
#pragma unroll
    for (int i = 0; i < 8; ++i) {
      ah[s][i]     = hrow[k0 + i];
      ah[s][i + 8] = hrow[k0 + 16 + i];
    }
  }

  // ---------- V = H @ W2 + b2 : 4 N-tiles ----------
#pragma unroll
  for (int nt = 0; nt < 4; ++nt) {
    v8f acc = {};
#pragma unroll
    for (int s = 0; s < 4; ++s) acc = wmma_bf16(ah[s], loadB(W2t, nt, s), acc);
    const float bias = b2[nt * 16 + l16];
#pragma unroll
    for (int v = 0; v < 8; ++v) {
      const int m = v + hi * 8;
      V[(size_t)(rowbase + m) * DDIM + nt * 16 + l16] = acc[v] + bias;
    }
  }
}

// ---------------- node init: deg=1 (self loop), m=-inf(enc), s=0 ----------------
__global__ void init_nodes_kernel(float* deg, unsigned* m_u, float* s, int n) {
  int i = blockIdx.x * blockDim.x + threadIdx.x;
  if (i < n) { deg[i] = 1.0f; m_u[i] = enc_f32(-3.402823466e38f); s[i] = 0.0f; }
}

__global__ void deg_kernel(const int* __restrict__ ei, float* deg, int E) {
  int e = blockIdx.x * blockDim.x + threadIdx.x;
  if (e < E) atomicAdd(&deg[ei[e]], 1.0f);   // ei[0..E) = row
}

__global__ void deg_fin_kernel(const float* __restrict__ deg,
                               float* dsq, float* dis, int n) {
  int i = blockIdx.x * blockDim.x + threadIdx.x;
  if (i < n) { float d = deg[i]; dsq[i] = sqrtf(d); dis[i] = rsqrtf(d); }
}

// ---------------- edge scores: 2 edges per wave, 16 lanes/edge ----------------
__global__ void score_kernel(const int* __restrict__ ei,
                             const float* __restrict__ Q,
                             const float* __restrict__ Kf,
                             float* __restrict__ score, unsigned* m_u,
                             int E, int Etot) {
  const int lane = threadIdx.x & 31;
  const int sub  = lane & 15;
  const int wv   = (blockIdx.x * blockDim.x + threadIdx.x) >> 5;
  const int e    = wv * 2 + (lane >> 4);
  if (e >= Etot) return;
  const int row = (e < E) ? ei[e]     : (e - E);
  const int col = (e < E) ? ei[E + e] : (e - E);
  const float4 qa = ((const float4*)(Q  + (size_t)row * AUDIM))[sub];
  const float4 ka = ((const float4*)(Kf + (size_t)col * AUDIM))[sub];
  float p = qa.x * ka.x + qa.y * ka.y + qa.z * ka.z + qa.w * ka.w;
#pragma unroll
  for (int off = 1; off < 16; off <<= 1) p += __shfl_xor(p, off);
  if (sub == 0) {
    const float sc = p * 0.125f;           // / sqrt(64)
    score[e] = sc;
    atomicMax(m_u + row, enc_f32(sc));
  }
}

__global__ void expsum_kernel(const int* __restrict__ ei,
                              const unsigned* __restrict__ m_u,
                              float* __restrict__ score, float* s,
                              int E, int Etot) {
  int e = blockIdx.x * blockDim.x + threadIdx.x;
  if (e >= Etot) return;
  const int row = (e < E) ? ei[e] : (e - E);
  const float es = __expf(score[e] - dec_f32(m_u[row]));
  score[e] = es;
  atomicAdd(&s[row], es);
}

__global__ void affinity_kernel(const int* __restrict__ ei,
                                const float* __restrict__ score,
                                const float* __restrict__ s,
                                const float* __restrict__ dsq,
                                const float* __restrict__ dis,
                                float* __restrict__ aff, int E, int Etot) {
  int e = blockIdx.x * blockDim.x + threadIdx.x;
  if (e >= Etot) return;
  const int row = (e < E) ? ei[e]     : (e - E);
  const int col = (e < E) ? ei[E + e] : (e - E);
  const float gat = dsq[row] * (score[e] / s[row]) * dis[col];
  const float gcn = dis[row] * dis[col];
  aff[e] = BETA * gcn + (1.0f - BETA) * gat;
}

// ---------------- hop weights: softmax(leaky_relu(att_logits)) ----------------
__global__ void hopw_kernel(const float* __restrict__ att, float* __restrict__ hw) {
  if (blockIdx.x == 0 && threadIdx.x == 0) {
    float lr[TITER], mx = -3.402823466e38f;
    for (int t = 0; t < TITER; ++t) {
      float v = att[t];
      lr[t] = (v > 0.0f) ? v : 0.2f * v;
      mx = fmaxf(mx, lr[t]);
    }
    float sum = 0.0f;
    for (int t = 0; t < TITER; ++t) { lr[t] = __expf(lr[t] - mx); sum += lr[t]; }
    const float inv = 1.0f / (sum * (1.0f + 1e-8f));
    for (int t = 0; t < TITER; ++t) hw[t] = lr[t] * inv;
  }
}

// ---------------- propagation init: h=V, agg=0, out=0 ----------------
__global__ void init_prop_kernel(const float* __restrict__ V, float* h,
                                 float* agg, float* out, int total) {
  int i = blockIdx.x * blockDim.x + threadIdx.x;
  if (i < total) { h[i] = V[i]; agg[i] = 0.0f; out[i] = 0.0f; }
}

// ---------------- push SpMM: 16 lanes/edge, float4 gather, f32 atomics ----------
__global__ void scatter_kernel(const int* __restrict__ ei,
                               const float* __restrict__ aff,
                               const float* __restrict__ h,
                               float* __restrict__ agg, int E, int Etot) {
  const int t   = blockIdx.x * blockDim.x + threadIdx.x;
  const int e   = t >> 4;
  const int sub = t & 15;
  if (e >= Etot) return;
  const int row = (e < E) ? ei[e]     : (e - E);
  const int col = (e < E) ? ei[E + e] : (e - E);
  const float a = aff[e];
  const float4 hv = ((const float4*)(h + (size_t)col * DDIM))[sub];
  float* dst = agg + (size_t)row * DDIM + sub * 4;
  atomicAdd(dst + 0, a * hv.x);
  atomicAdd(dst + 1, a * hv.y);
  atomicAdd(dst + 2, a * hv.z);
  atomicAdd(dst + 3, a * hv.w);
}

// h = agg*(1-a) + V*a; out += hopw[t]*h; agg reset for next iteration
__global__ void update_kernel(float* __restrict__ h, float* __restrict__ agg,
                              const float* __restrict__ V, float* __restrict__ out,
                              const float* __restrict__ hw, int t, int total) {
  int i = blockIdx.x * blockDim.x + threadIdx.x;
  if (i < total) {
    const float hn = agg[i] * (1.0f - ALPHA) + V[i] * ALPHA;
    h[i]   = hn;
    agg[i] = 0.0f;
    out[i] += hw[t] * hn;
  }
}

// ---------------- host launch ----------------
extern "C" void kernel_launch(void* const* d_in, const int* in_sizes, int n_in,
                              void* d_out, int out_size, void* d_ws, size_t ws_size,
                              hipStream_t stream) {
  const float* x   = (const float*)d_in[0];
  const float* Wq  = (const float*)d_in[1];
  const float* bq  = (const float*)d_in[2];
  const float* Wk  = (const float*)d_in[3];
  const float* bk  = (const float*)d_in[4];
  const float* W1  = (const float*)d_in[5];
  const float* b1  = (const float*)d_in[6];
  const float* W2  = (const float*)d_in[7];
  const float* b2  = (const float*)d_in[8];
  const float* att = (const float*)d_in[9];
  const int*   ei  = (const int*)d_in[10];
  float* out = (float*)d_out;

  // workspace carve-out (256B aligned)
  char* w = (char*)d_ws;
  size_t off = 0;
  auto carve = [&](size_t bytes) -> void* {
    void* p = w + off;
    off = (off + bytes + 255) & ~(size_t)255;
    return p;
  };
  __bf16* W1t = (__bf16*)carve((size_t)FDIM * H1DIM * 2);
  __bf16* W2t = (__bf16*)carve((size_t)H1DIM * DDIM * 2);
  __bf16* Wqt = (__bf16*)carve((size_t)FDIM * AUDIM * 2);
  __bf16* Wkt = (__bf16*)carve((size_t)FDIM * AUDIM * 2);
  float* Q    = (float*)carve((size_t)NNODES * AUDIM * 4);
  float* Kf   = (float*)carve((size_t)NNODES * AUDIM * 4);
  float* V    = (float*)carve((size_t)NNODES * DDIM * 4);
  float* h    = (float*)carve((size_t)NNODES * DDIM * 4);
  float* agg  = (float*)carve((size_t)NNODES * DDIM * 4);
  float* deg  = (float*)carve((size_t)NNODES * 4);
  float* dsq  = (float*)carve((size_t)NNODES * 4);
  float* dis  = (float*)carve((size_t)NNODES * 4);
  unsigned* m_u = (unsigned*)carve((size_t)NNODES * 4);
  float* ssum = (float*)carve((size_t)NNODES * 4);
  float* score = (float*)carve((size_t)ETOT * 4);
  float* aff   = (float*)carve((size_t)ETOT * 4);
  float* hw    = (float*)carve((size_t)TITER * 4);

  const int TB = 256;

  // 1) weight transpose + bf16 convert
  wprep_kernel<<<(FDIM * H1DIM + TB - 1) / TB, TB, 0, stream>>>(W1, W1t, FDIM, H1DIM);
  wprep_kernel<<<(H1DIM * DDIM + TB - 1) / TB, TB, 0, stream>>>(W2, W2t, H1DIM, DDIM);
  wprep_kernel<<<(FDIM * AUDIM + TB - 1) / TB, TB, 0, stream>>>(Wq, Wqt, FDIM, AUDIM);
  wprep_kernel<<<(FDIM * AUDIM + TB - 1) / TB, TB, 0, stream>>>(Wk, Wkt, FDIM, AUDIM);

  // 2) fused WMMA node MLP -> Q, K, V
  const int numTiles = (NNODES + 15) / 16;            // 6250
  node_mlp_kernel<<<(numTiles + 7) / 8, TB, 0, stream>>>(
      x, W1t, b1, W2t, b2, Wqt, bq, Wkt, bk, Q, Kf, V, numTiles);

  // 3) degrees + softmax state
  init_nodes_kernel<<<(NNODES + TB - 1) / TB, TB, 0, stream>>>(deg, m_u, ssum, NNODES);
  deg_kernel<<<(NEDGES + TB - 1) / TB, TB, 0, stream>>>(ei, deg, NEDGES);
  deg_fin_kernel<<<(NNODES + TB - 1) / TB, TB, 0, stream>>>(deg, dsq, dis, NNODES);

  // 4) edge attention: score + segment max, exp + segment sum, affinity
  {
    const int waves = (ETOT + 1) / 2;
    const long long threads = (long long)waves * 32;
    score_kernel<<<(int)((threads + TB - 1) / TB), TB, 0, stream>>>(
        ei, Q, Kf, score, m_u, NEDGES, ETOT);
  }
  expsum_kernel<<<(ETOT + TB - 1) / TB, TB, 0, stream>>>(ei, m_u, score, ssum, NEDGES, ETOT);
  affinity_kernel<<<(ETOT + TB - 1) / TB, TB, 0, stream>>>(ei, score, ssum, dsq, dis,
                                                           aff, NEDGES, ETOT);

  // 5) hop weights
  hopw_kernel<<<1, 32, 0, stream>>>(att, hw);

  // 6) propagation: T iterations of push-SpMM + update (L2-resident working set)
  const int total = NNODES * DDIM;
  init_prop_kernel<<<(total + TB - 1) / TB, TB, 0, stream>>>(V, h, agg, out, total);
  for (int t = 0; t < TITER; ++t) {
    const long long sthreads = (long long)ETOT * 16;
    scatter_kernel<<<(int)((sthreads + TB - 1) / TB), TB, 0, stream>>>(
        ei, aff, h, agg, NEDGES, ETOT);
    update_kernel<<<(total + TB - 1) / TB, TB, 0, stream>>>(h, agg, V, out, hw, t, total);
  }
}
